// RPN_39187281609223
// MI455X (gfx1250) — matrix operand order repaired and checked by reading, not compiled, not run on hardware
//
#include <hip/hip_runtime.h>
#include <hip/hip_bf16.h>

typedef __attribute__((ext_vector_type(16))) _Float16 v16h;
typedef __attribute__((ext_vector_type(8)))  _Float16 v8h;
typedef __attribute__((ext_vector_type(8)))  float    v8f;

#define RPN_B 8
#define RPN_C 256
#define RPN_H 100
#define RPN_W 100
#define RPN_A 9
#define NPOS  80000          // B*H*W
#define KSTEPS 72            // 2304/32
#define PRE_NMS 2000
#define POST_NMS 300
#define NEGV (-1.0e9f)

// ---------------------------------------------------------------------------
// WMMA A-fragment swizzle (16x32 f16 MxK, ISA 7.12.2):
// lane L: M = L&15, g = L>>4; v16h elem h -> K = (h<8 ? 8g+h : 16+8g+(h-8)).
// ---------------------------------------------------------------------------
__device__ __forceinline__ int kloc_of(int h, int g) {
  return (h < 8) ? (8 * g + h) : (16 + 8 * g + (h - 8));
}

// Kernel 0a: pre-swizzle 3x3 conv weights. Layout:
// wfrag[((s*16 + ct)*32 + lane)*16 + h]; k = tap*256 + ci.
__global__ void prep_wfrag(const float* __restrict__ wconv,
                           _Float16* __restrict__ wfrag) {
  int idx = blockIdx.x * blockDim.x + threadIdx.x;
  if (idx >= KSTEPS * 16 * 32 * 16) return;
  int h  = idx & 15;
  int L  = (idx >> 4) & 31;
  int ct = (idx >> 9) & 15;
  int s  = idx >> 13;
  int co = ct * 16 + (L & 15);
  int k  = s * 32 + 16 * (L >> 4) + h;
  int tap = k >> 8, ci = k & 255;
  wfrag[idx] = (_Float16)wconv[((size_t)co * 256 + ci) * 9 + tap];
}

// Kernel 0b: pre-swizzle 1x1 head weights into 48x256 (rows: 9 cls, 36 reg, 3 pad).
// wfragH[((s*3 + ct)*32 + lane)*16 + h], s = K-step (8), ct = 16-row tile (3).
__global__ void prep_wfragH(const float* __restrict__ wcls,
                            const float* __restrict__ wreg,
                            _Float16* __restrict__ wfragH) {
  int idx = blockIdx.x * blockDim.x + threadIdx.x;
  if (idx >= 8 * 3 * 32 * 16) return;
  int h  = idx & 15;
  int L  = (idx >> 4) & 31;
  int sc = idx >> 9;            // s*3 + ct
  int ct = sc % 3, s = sc / 3;
  int co = ct * 16 + (L & 15);
  int k  = s * 32 + 16 * (L >> 4) + h;
  float v = 0.f;
  if (co < 9)       v = wcls[(size_t)co * 256 + k];
  else if (co < 45) v = wreg[(size_t)(co - 9) * 256 + k];
  wfragH[idx] = (_Float16)v;
}

// ---------------------------------------------------------------------------
// Kernel 1: implicit-GEMM 3x3 conv + bias + ReLU.
// Block = 256 threads (8 waves) -> 32 positions x 256 output channels.
// Wave owns 32 co (2 A fragments); 2 B fragments (n 0-15 / 16-31) in LDS
// -> 4 v_wmma_f32_16x16x32_f16 per K-step. tap hoisted; branchless im2col.
// ---------------------------------------------------------------------------
__global__ __launch_bounds__(256) void conv3x3_wmma(
    const float* __restrict__ feats, const float* __restrict__ bconv,
    const _Float16* __restrict__ wfrag, _Float16* __restrict__ tout) {
  __shared__ __align__(16) _Float16 Bl[32][40];   // [n][k], padded rows
  __shared__ __align__(16) _Float16 T[32][264];   // [n][co], padded rows

  const int tid  = threadIdx.x;
  const int lane = tid & 31;
  const int wv   = tid >> 5;
  const int g    = lane >> 4;
  const int m    = lane & 15;
  const int p0   = blockIdx.x * 32;

  // im2col fill assignment: thread -> position n = tid&31, klocs {kb,kb+8,kb+16,kb+24}
  const int nfill = tid & 31;
  const int kb    = tid >> 5;
  const int pfill = p0 + nfill;
  const int bimg  = pfill / (RPN_H * RPN_W);
  const int rem   = pfill - bimg * (RPN_H * RPN_W);
  const int hpos  = rem / RPN_W;
  const int wpos  = rem - hpos * RPN_W;
  const float* fB = feats + (size_t)bimg * RPN_C * (RPN_H * RPN_W);

  const _Float16* pa0 = wfrag + ((size_t)(2 * wv) * 32 + lane) * 16;
  const _Float16* pa1 = pa0 + 512;   // next 16-row tile

  v8f c00 = {}, c01 = {}, c10 = {}, c11 = {};

  int s = 0;
  for (int tap = 0; tap < 9; ++tap) {
    const int dh = tap / 3 - 1, dw = tap % 3 - 1;
    const int hh = hpos + dh, ww = wpos + dw;
    const bool inb = ((unsigned)hh < (unsigned)RPN_H) &&
                     ((unsigned)ww < (unsigned)RPN_W);
    const int hc = min(max(hh, 0), RPN_H - 1);
    const int wc = min(max(ww, 0), RPN_W - 1);
    const float* fp = fB + (size_t)hc * RPN_W + wc + (size_t)kb * (RPN_H * RPN_W);
    for (int cc = 0; cc < 8; ++cc, ++s) {
      __syncthreads();
      // branchless clamped loads, zeroed by select when out of image
      float v0 = fp[0];
      float v1 = fp[8  * (RPN_H * RPN_W)];
      float v2 = fp[16 * (RPN_H * RPN_W)];
      float v3 = fp[24 * (RPN_H * RPN_W)];
      if (!inb) { v0 = 0.f; v1 = 0.f; v2 = 0.f; v3 = 0.f; }
      Bl[nfill][kb]      = (_Float16)v0;
      Bl[nfill][kb + 8]  = (_Float16)v1;
      Bl[nfill][kb + 16] = (_Float16)v2;
      Bl[nfill][kb + 24] = (_Float16)v3;
      fp += 32 * (RPN_H * RPN_W);
      __syncthreads();

      v16h a0 = *(const v16h*)(pa0 + (size_t)s * 8192);
      v16h a1 = *(const v16h*)(pa1 + (size_t)s * 8192);
      v8h b0l = *(const v8h*)&Bl[m][16 * g];
      v8h b0h = *(const v8h*)&Bl[m][16 * g + 8];
      v8h b1l = *(const v8h*)&Bl[m + 16][16 * g];
      v8h b1h = *(const v8h*)&Bl[m + 16][16 * g + 8];
      v16h bf0, bf1;
#pragma unroll
      for (int i = 0; i < 8; ++i) {
        bf0[i] = b0l[i]; bf0[8 + i] = b0h[i];
        bf1[i] = b1l[i]; bf1[8 + i] = b1h[i];
      }
      c00 = __builtin_amdgcn_wmma_f32_16x16x32_f16(false, a0, false, bf0,
                                                   (short)0, c00, false, false);
      c01 = __builtin_amdgcn_wmma_f32_16x16x32_f16(false, a0, false, bf1,
                                                   (short)0, c01, false, false);
      c10 = __builtin_amdgcn_wmma_f32_16x16x32_f16(false, a1, false, bf0,
                                                   (short)0, c10, false, false);
      c11 = __builtin_amdgcn_wmma_f32_16x16x32_f16(false, a1, false, bf1,
                                                   (short)0, c11, false, false);
    }
  }

  // epilogue: bias + ReLU, LDS transpose, coalesced NHWC f16 store
  __syncthreads();
  {
    const int ct0 = 2 * wv;
#pragma unroll
    for (int v = 0; v < 8; ++v) {
      const int co0 = ct0 * 16 + v + 8 * g;
      const int co1 = co0 + 16;
      const float bi0 = bconv[co0], bi1 = bconv[co1];
      float r;
      r = c00[v] + bi0; T[m][co0]      = (_Float16)(r > 0.f ? r : 0.f);
      r = c01[v] + bi0; T[m + 16][co0] = (_Float16)(r > 0.f ? r : 0.f);
      r = c10[v] + bi1; T[m][co1]      = (_Float16)(r > 0.f ? r : 0.f);
      r = c11[v] + bi1; T[m + 16][co1] = (_Float16)(r > 0.f ? r : 0.f);
    }
  }
  __syncthreads();
  {
    const int n   = tid >> 3;
    const int seg = (tid & 7) * 32;
    const uint4* src = (const uint4*)&T[n][seg];
    uint4* dst = (uint4*)&tout[(size_t)(p0 + n) * 256 + seg];
    dst[0] = src[0]; dst[1] = src[1]; dst[2] = src[2]; dst[3] = src[3];
  }
}

// ---------------------------------------------------------------------------
// Kernel 2: 1x1 heads as WMMA GEMM (48x16x256 per wave; rows 0-8 cls,
// 9-44 reg, 45-47 pad) + sigmoid + delta decode + clip + area filter.
// B fragments load directly from NHWC f16 activations (no LDS staging).
// Block = 8 waves = 128 positions.
// ---------------------------------------------------------------------------
__global__ __launch_bounds__(256) void heads_wmma(
    const _Float16* __restrict__ t, const _Float16* __restrict__ wfragH,
    const float* __restrict__ bcls, const float* __restrict__ breg,
    float* __restrict__ scores, float* __restrict__ props) {
  __shared__ float TL[8][16][49];   // [wave][n][co], padded to kill conflicts

  const int tid  = threadIdx.x;
  const int lane = tid & 31;
  const int wv   = tid >> 5;
  const int g    = lane >> 4;
  const int m    = lane & 15;
  const int p0   = blockIdx.x * 128 + wv * 16;

  v8f c0 = {}, c1 = {}, c2 = {};
#pragma unroll
  for (int s = 0; s < 8; ++s) {
    v16h bf = *(const v16h*)&t[(size_t)(p0 + m) * 256 + s * 32 + 16 * g];
    v16h a0 = *(const v16h*)&wfragH[((size_t)(s * 3 + 0) * 32 + lane) * 16];
    v16h a1 = *(const v16h*)&wfragH[((size_t)(s * 3 + 1) * 32 + lane) * 16];
    v16h a2 = *(const v16h*)&wfragH[((size_t)(s * 3 + 2) * 32 + lane) * 16];
    c0 = __builtin_amdgcn_wmma_f32_16x16x32_f16(false, a0, false, bf,
                                                (short)0, c0, false, false);
    c1 = __builtin_amdgcn_wmma_f32_16x16x32_f16(false, a1, false, bf,
                                                (short)0, c1, false, false);
    c2 = __builtin_amdgcn_wmma_f32_16x16x32_f16(false, a2, false, bf,
                                                (short)0, c2, false, false);
  }
#pragma unroll
  for (int v = 0; v < 8; ++v) {
    const int r = v + 8 * g;
    TL[wv][m][r]      = c0[v];
    TL[wv][m][16 + r] = c1[v];
    TL[wv][m][32 + r] = c2[v];
  }
  __syncthreads();

  if (tid >= 128) return;
  const int p = blockIdx.x * 128 + tid;
  const int w2 = tid >> 4, n2 = tid & 15;
  float acc[45];
#pragma unroll
  for (int j = 0; j < 45; ++j) acc[j] = TL[w2][n2][j];

  const int b  = p / (RPN_H * RPN_W);
  const int rm = p - b * (RPN_H * RPN_W);
  const int h  = rm / RPN_W;
  const int w  = rm - h * RPN_W;
  const float cx = (w + 0.5f) * 16.f;
  const float cy = (h + 0.5f) * 16.f;
  float* sc_out = scores + (size_t)b * (RPN_H * RPN_W * RPN_A) +
                  (size_t)(h * RPN_W + w) * RPN_A;
  float* pr_out = props + ((size_t)b * (RPN_H * RPN_W * RPN_A) +
                           (size_t)(h * RPN_W + w) * RPN_A) * 4;
#pragma unroll
  for (int a = 0; a < RPN_A; ++a) {
    const int ir = a / 3, is = a - ir * 3;
    const float r  = (ir == 0) ? 0.5f : (ir == 1 ? 1.f : 2.f);
    const float sz = (is == 0) ? 128.f : (is == 1 ? 256.f : 512.f);
    const float sq = sqrtf(r);
    const float aw = sz / sq, ah = sz * sq;
    const float sg = 1.f / (1.f + expf(-(acc[a] + bcls[a])));
    const float dx  = acc[9 + a * 4 + 0] + breg[a * 4 + 0];
    const float dy  = acc[9 + a * 4 + 1] + breg[a * 4 + 1];
    const float dwv = fminf(acc[9 + a * 4 + 2] + breg[a * 4 + 2], 4.135f);
    const float dhv = fminf(acc[9 + a * 4 + 3] + breg[a * 4 + 3], 4.135f);
    const float pcx = dx * aw + cx;
    const float pcy = dy * ah + cy;
    const float pw  = expf(dwv) * aw;
    const float ph  = expf(dhv) * ah;
    const float x1 = fminf(fmaxf(pcx - 0.5f * pw, 0.f), (float)(RPN_W * 16));
    const float y1 = fminf(fmaxf(pcy - 0.5f * ph, 0.f), (float)(RPN_H * 16));
    const float x2 = fminf(fmaxf(pcx + 0.5f * pw, 0.f), (float)(RPN_W * 16));
    const float y2 = fminf(fmaxf(pcy + 0.5f * ph, 0.f), (float)(RPN_H * 16));
    const float area = (x2 - x1) * (y2 - y1);
    sc_out[a] = (area >= 25.f) ? sg : NEGV;
    pr_out[a * 4 + 0] = x1; pr_out[a * 4 + 1] = y1;
    pr_out[a * 4 + 2] = x2; pr_out[a * 4 + 3] = y2;
  }
}

// ---------------------------------------------------------------------------
// Kernel 3: exact top-2000 per image: 4-pass radix select + atomic compaction.
// ---------------------------------------------------------------------------
__device__ __forceinline__ unsigned fkey(float s) {
  unsigned u = __float_as_uint(s);
  return (u & 0x80000000u) ? ~u : (u | 0x80000000u);
}

__global__ __launch_bounds__(1024) void topk2000(
    const float* __restrict__ scores, const float* __restrict__ props,
    float* __restrict__ tsc, float* __restrict__ tbox) {
  __shared__ unsigned hist[256];
  __shared__ unsigned sPrefix, sR, cGt, cEq;
  const int b = blockIdx.x, tid = threadIdx.x;
  const int N = RPN_H * RPN_W * RPN_A;
  const float* sc = scores + (size_t)b * N;

  unsigned prefix = 0, R = PRE_NMS;
  for (int pass = 0; pass < 4; ++pass) {
    const int shift = 24 - 8 * pass;
    const unsigned pmask = (pass == 0) ? 0u : (0xFFFFFFFFu << (shift + 8));
    if (tid < 256) hist[tid] = 0;
    __syncthreads();
    for (int i = tid; i < N; i += 1024) {
      unsigned key = fkey(sc[i]);
      if ((key & pmask) == (prefix & pmask))
        atomicAdd(&hist[(key >> shift) & 255], 1u);
    }
    __syncthreads();
    if (tid == 0) {
      unsigned cum = 0; int v = 255;
      for (; v > 0; --v) {
        if (cum + hist[v] >= R) break;
        cum += hist[v];
      }
      sPrefix = prefix | ((unsigned)v << shift);
      sR = R - cum;
    }
    __syncthreads();
    prefix = sPrefix; R = sR;
    __syncthreads();
  }
  const unsigned Kth = prefix;
  const unsigned nGt = PRE_NMS - R;
  if (tid == 0) { cGt = 0; cEq = 0; }
  __syncthreads();
  const float* pb = props + (size_t)b * N * 4;
  for (int i = tid; i < N; i += 1024) {
    float s = sc[i];
    unsigned key = fkey(s);
    int slot = -1;
    if (key > Kth) {
      slot = (int)atomicAdd(&cGt, 1u);
    } else if (key == Kth) {
      unsigned e = atomicAdd(&cEq, 1u);
      if (e < R) slot = (int)(nGt + e);
    }
    if (slot >= 0) {
      tsc[(size_t)b * PRE_NMS + slot] = s;
      float4 bx = *(const float4*)(pb + (size_t)i * 4);
      *(float4*)(tbox + ((size_t)b * PRE_NMS + slot) * 4) = bx;
    }
  }
}

// ---------------------------------------------------------------------------
// Kernel 4: greedy NMS, 300 iterations, one block per image, LDS resident.
// out: boxes [B,300,4] then scores [B,300] (tuple return order).
// ---------------------------------------------------------------------------
__global__ __launch_bounds__(1024) void nms300(
    const float* __restrict__ tsc, const float* __restrict__ tbox,
    float* __restrict__ out) {
  __shared__ float S[PRE_NMS], X1[PRE_NMS], Y1[PRE_NMS];
  __shared__ float X2[PRE_NMS], Y2[PRE_NMS], AR[PRE_NMS];
  __shared__ float wmax[32]; __shared__ int widx[32];
  __shared__ float bb[5]; __shared__ int bestI;
  const int b = blockIdx.x, tid = threadIdx.x;
  for (int i = tid; i < PRE_NMS; i += 1024) {
    float4 box = *(const float4*)(tbox + ((size_t)b * PRE_NMS + i) * 4);
    S[i] = tsc[(size_t)b * PRE_NMS + i];
    X1[i] = box.x; Y1[i] = box.y; X2[i] = box.z; Y2[i] = box.w;
    AR[i] = (box.z - box.x) * (box.w - box.y);
  }
  __syncthreads();
  float* oBox = out + (size_t)b * POST_NMS * 4;
  float* oSc  = out + (size_t)RPN_B * POST_NMS * 4 + (size_t)b * POST_NMS;

  for (int it = 0; it < POST_NMS; ++it) {
    float ms = -__builtin_inff(); int mi = 0;
    for (int i = tid; i < PRE_NMS; i += 1024)
      if (S[i] > ms) { ms = S[i]; mi = i; }
    for (int off = 16; off > 0; off >>= 1) {
      float os = __shfl_down(ms, off, 32);
      int   oi = __shfl_down(mi, off, 32);
      if (os > ms || (os == ms && oi < mi)) { ms = os; mi = oi; }
    }
    if ((tid & 31) == 0) { wmax[tid >> 5] = ms; widx[tid >> 5] = mi; }
    __syncthreads();
    if (tid < 32) {
      ms = wmax[tid]; mi = widx[tid];
      for (int off = 16; off > 0; off >>= 1) {
        float os = __shfl_down(ms, off, 32);
        int   oi = __shfl_down(mi, off, 32);
        if (os > ms || (os == ms && oi < mi)) { ms = os; mi = oi; }
      }
      if (tid == 0) {
        bestI = mi;
        bb[0] = X1[mi]; bb[1] = Y1[mi]; bb[2] = X2[mi]; bb[3] = Y2[mi];
        bb[4] = AR[mi];
        const bool valid = ms > 0.5f * NEGV;
        oBox[it * 4 + 0] = valid ? bb[0] : 0.f;
        oBox[it * 4 + 1] = valid ? bb[1] : 0.f;
        oBox[it * 4 + 2] = valid ? bb[2] : 0.f;
        oBox[it * 4 + 3] = valid ? bb[3] : 0.f;
        oSc[it] = valid ? ms : -1.f;
      }
    }
    __syncthreads();
    const int mbest = bestI;
    const float bx1 = bb[0], by1 = bb[1], bx2 = bb[2], by2 = bb[3], ba = bb[4];
    for (int i = tid; i < PRE_NMS; i += 1024) {
      float xx1 = fmaxf(bx1, X1[i]);
      float yy1 = fmaxf(by1, Y1[i]);
      float xx2 = fminf(bx2, X2[i]);
      float yy2 = fminf(by2, Y2[i]);
      float inter = fmaxf(xx2 - xx1, 0.f) * fmaxf(yy2 - yy1, 0.f);
      float iou = inter / (ba + AR[i] - inter + 1e-9f);
      if (iou > 0.7f || i == mbest) S[i] = NEGV;
    }
    __syncthreads();
  }
}

// ---------------------------------------------------------------------------
extern "C" void kernel_launch(void* const* d_in, const int* in_sizes, int n_in,
                              void* d_out, int out_size, void* d_ws,
                              size_t ws_size, hipStream_t stream) {
  (void)in_sizes; (void)n_in; (void)out_size; (void)ws_size;
  const float* feats  = (const float*)d_in[0];
  const float* w_conv = (const float*)d_in[1];
  const float* b_conv = (const float*)d_in[2];
  const float* w_cls  = (const float*)d_in[3];
  const float* b_cls  = (const float*)d_in[4];
  const float* w_reg  = (const float*)d_in[5];
  const float* b_reg  = (const float*)d_in[6];

  char* ws = (char*)d_ws;
  _Float16* wfrag  = (_Float16*)(ws + 0);          //  1,179,648 B
  _Float16* tbuf   = (_Float16*)(ws + 1179648);    // 40,960,000 B
  float* scores    = (float*)(ws + 42139648);      //  2,880,000 B
  float* props     = (float*)(ws + 45019648);      // 11,520,000 B
  float* tsc       = (float*)(ws + 56539648);      //     64,000 B
  float* tbox      = (float*)(ws + 56603648);      //    256,000 B
  _Float16* wfragH = (_Float16*)(ws + 56859648);   //     24,576 B

  prep_wfrag<<<(KSTEPS * 16 * 32 * 16 + 255) / 256, 256, 0, stream>>>(w_conv, wfrag);
  prep_wfragH<<<(8 * 3 * 32 * 16 + 255) / 256, 256, 0, stream>>>(w_cls, w_reg, wfragH);
  conv3x3_wmma<<<NPOS / 32, 256, 0, stream>>>(feats, b_conv, wfrag, tbuf);
  heads_wmma<<<NPOS / 128, 256, 0, stream>>>(tbuf, wfragH, b_cls, b_reg, scores, props);
  topk2000<<<RPN_B, 1024, 0, stream>>>(scores, props, tsc, tbox);
  nms300<<<RPN_B, 1024, 0, stream>>>(tsc, tbox, (float*)d_out);
}
